// HedgehogBlock_81716047774353
// MI455X (gfx1250) — matrix operand order repaired and checked by reading, not compiled, not run on hardware
//
#include <hip/hip_runtime.h>
#include <hip/hip_bf16.h>

// ---------------- CDNA5 WMMA types / helpers (wave32) ----------------
typedef __attribute__((ext_vector_type(16))) _Float16 v16h;
typedef __attribute__((ext_vector_type(8)))  _Float16 v8h;
typedef __attribute__((ext_vector_type(4)))  _Float16 v4h;
typedef __attribute__((ext_vector_type(8)))  float    v8f;
typedef __attribute__((ext_vector_type(4)))  int      v4i;

#define B_ 4
#define N_ 4096
#define D_ 1024
#define H_ 16
#define DH_ 64
#define EPS_ 1e-6f

#if __has_builtin(__builtin_amdgcn_global_load_async_to_lds_b128)
#define HAS_ASYNC_LDS 1
#else
#define HAS_ASYNC_LDS 0
#endif

__device__ __forceinline__ v8f wmma16(v16h a, v16h b, v8f c) {
  // D = A(16x32 f16) * B(32x16 f16) + C(16x16 f32)
  return __builtin_amdgcn_wmma_f32_16x16x32_f16(
      /*neg_a=*/false, a, /*neg_b=*/false, b,
      /*c_mod=*/(short)0, c, /*reuse_a=*/false, /*reuse_b=*/false);
}

__device__ __forceinline__ v16h cat16(v8h lo, v8h hi) {
  v16h r;
#pragma unroll
  for (int j = 0; j < 8; ++j) { r[j] = lo[j]; r[j + 8] = hi[j]; }
  return r;
}

// A fragment (16x32), source stored row-major [m][k], ld % 8 == 0.
// ISA layout: lanes 0-15 -> M=lane, K {kb..kb+7, 16+kb..16+kb+7}, kb = 8*(lane>=16).
__device__ __forceinline__ v16h load_a_rm(const _Float16* p, int row0, int k0, int ld) {
  const int lane = threadIdx.x & 31;
  const int m  = row0 + (lane & 15);
  const int kb = (lane < 16) ? 0 : 8;
  const _Float16* base = p + m * ld + k0 + kb;
  v8h lo = *(const v8h*)(base);
  v8h hi = *(const v8h*)(base + 16);
  return cat16(lo, hi);
}

// B fragment (32x16), source stored n-major [n][k], ld % 8 == 0.
// ISA layout: lanes 0-15 -> N=lane, K {kb..kb+7, 16+kb..16+kb+7}, kb = 8*(lane>=16).
__device__ __forceinline__ v16h load_b_nm(const _Float16* p, int col0, int k0, int ld) {
  const int lane = threadIdx.x & 31;
  const int n  = col0 + (lane & 15);
  const int kb = (lane < 16) ? 0 : 8;
  const _Float16* base = p + n * ld + k0 + kb;
  v8h lo = *(const v8h*)(base);
  v8h hi = *(const v8h*)(base + 16);
  return cat16(lo, hi);
}

// Stage a 64-row x 32-half f16 tile from global into LDS (ld = 40 halves = 80B, 16B-aligned
// rows). 128 threads, 256 16-byte chunks -> 2 per thread. Async global->LDS when available.
__device__ __forceinline__ void stage_tile(const _Float16* __restrict__ g, size_t gld,
                                           _Float16* lds, int k0) {
  const int tid = threadIdx.x;
#pragma unroll
  for (int ci = 0; ci < 2; ++ci) {
    const int c = tid + ci * 128;      // 0..255
    const int row = c >> 2, seg = c & 3;
    const _Float16* gp = g + (size_t)row * gld + k0 + seg * 8;
    _Float16* lp = lds + row * 40 + seg * 8;
#if HAS_ASYNC_LDS
    // flat->global: identical bits; flat->LDS: low 32 bits are the LDS offset (ISA aperture rule)
    __builtin_amdgcn_global_load_async_to_lds_b128(
        (__attribute__((address_space(1))) v4i*)(uintptr_t)gp,
        (__attribute__((address_space(3))) v4i*)(unsigned)(uintptr_t)lp, 0, 0);
#else
    *(v8h*)lp = *(const v8h*)gp;
#endif
  }
}

__device__ __forceinline__ void wait_stage() {
#if HAS_ASYNC_LDS
#if __has_builtin(__builtin_amdgcn_s_wait_asynccnt)
  __builtin_amdgcn_s_wait_asynccnt(0);
#else
  asm volatile("s_wait_asynccnt 0x0" ::: "memory");
#endif
#endif
}

// ---------------- Kernel: f32 -> f16 conversion (x, Wq, Wk, Wv), 4-wide ----------------
__global__ __launch_bounds__(256) void hh_cvt(const float* __restrict__ s,
                                              _Float16* __restrict__ d, int n4) {
  const int i = (blockIdx.x * 256 + threadIdx.x) * 4;
  if (i < n4) {
    const float4 v = *(const float4*)(s + i);
    v4h o; o[0] = (_Float16)v.x; o[1] = (_Float16)v.y;
    o[2] = (_Float16)v.z; o[3] = (_Float16)v.w;
    *(v4h*)(d + i) = o;
  }
}

// ---------------- Kernel: theta = exp(W^T + W) per head, f16 (symmetric) ----------------
__global__ __launch_bounds__(256) void hh_theta(const float* __restrict__ w,
                                                _Float16* __restrict__ th) {
  const int h = blockIdx.x;
  const float* wh = w + (size_t)h * DH_ * DH_;
  for (int i = threadIdx.x; i < DH_ * DH_; i += 256) {
    const int d = i >> 6, e = i & 63;
    th[(size_t)h * DH_ * DH_ + i] = (_Float16)expf(wh[e * DH_ + d] + wh[d * DH_ + e]);
  }
}

// ---------------- Kernel: zero kv/ksum accumulators ----------------
__global__ __launch_bounds__(256) void hh_zero(float* __restrict__ p, int n) {
  const int i = blockIdx.x * 256 + threadIdx.x;
  if (i < n) p[i] = 0.0f;
}

// ---------------- Kernel: fused QKV + feature map + partial KV ----------------
// grid: (N/64, B*H), block: 128 threads = 4 waves; wave w owns M-strip [w*16, w*16+16)
__global__ __launch_bounds__(128) void hh_pass1(
    const _Float16* __restrict__ x16,                                   // [B][N][D] f16
    const _Float16* __restrict__ wq16, const float* __restrict__ bq,    // [D][D] f16
    const _Float16* __restrict__ wk16, const float* __restrict__ bk,
    const _Float16* __restrict__ wv16, const float* __restrict__ bv,
    const _Float16* __restrict__ theta,   // [H][DH][DH] f16, symmetric per head
    _Float16* __restrict__ qf_out,        // [B*H][N][DH] f16
    float* __restrict__ kv,               // [B*H][DH][DH]
    float* __restrict__ ksum)             // [B*H][DH]
{
  constexpr int LDW = 40;   // K-tile leading dim (32 + 8 pad)
  constexpr int LDS2 = 72;  // 64-wide staging leading dim (64 + 8 pad)
  constexpr int NK = D_ / 32;

  __shared__ __align__(16) _Float16 th[DH_ * DH_];   // [d][e]; symmetric -> n-major OK
  __shared__ __align__(16) _Float16 xs [2][64 * LDW];  // double-buffered K tiles
  __shared__ __align__(16) _Float16 wqs[2][64 * LDW];
  __shared__ __align__(16) _Float16 wks[2][64 * LDW];
  __shared__ __align__(16) _Float16 wvs[2][64 * LDW];
  __shared__ __align__(16) _Float16 qs  [64 * LDS2]; // [row][col]
  __shared__ __align__(16) _Float16 ks  [64 * LDS2]; // [row][col]
  __shared__ __align__(16) _Float16 vsT [64 * LDS2]; // [col e][row n]  (transposed)
  __shared__ __align__(16) _Float16 kfsT[64 * LDS2]; // [col d][row n]  (transposed)

  const int tid  = threadIdx.x;
  const int wave = tid >> 5;
  const int lane = tid & 31;
  const int bh = blockIdx.y;
  const int h  = bh & (H_ - 1);
  const int b  = bh >> 4;
  const int n0 = blockIdx.x * 64;

  for (int i = tid; i < DH_ * DH_; i += 128) th[i] = theta[(size_t)h * DH_ * DH_ + i];

  const _Float16* xrow  = x16  + ((size_t)b * N_ + n0) * D_;
  const _Float16* wqrow = wq16 + (size_t)(h * DH_) * D_;
  const _Float16* wkrow = wk16 + (size_t)(h * DH_) * D_;
  const _Float16* wvrow = wv16 + (size_t)(h * DH_) * D_;

  v8f accq[4], acck[4], accv[4];
#pragma unroll
  for (int t = 0; t < 4; ++t) { accq[t] = {}; acck[t] = {}; accv[t] = {}; }

  // ---- GEMM 1: q|k|v[64x64] = x[64x1024] * W_h^T, K chunked by 32 ----
  // Double-buffered async pipeline: copies for tile i+1 overlap WMMAs on tile i.
  stage_tile(xrow,  D_, xs [0], 0);
  stage_tile(wqrow, D_, wqs[0], 0);
  stage_tile(wkrow, D_, wks[0], 0);
  stage_tile(wvrow, D_, wvs[0], 0);
  for (int i = 0; i < NK; ++i) {
    const int cur = i & 1;
    wait_stage();            // copies into buffer `cur` (issued last iter) complete
    __syncthreads();         // publish; also: all waves done reading buffer `1-cur`
    if (i + 1 < NK) {
      const int k1 = (i + 1) * 32;
      stage_tile(xrow,  D_, xs [1 - cur], k1);
      stage_tile(wqrow, D_, wqs[1 - cur], k1);
      stage_tile(wkrow, D_, wks[1 - cur], k1);
      stage_tile(wvrow, D_, wvs[1 - cur], k1);
      if (i + 2 < NK) {      // prefetch tile i+2 into L2 (i+1 already in flight)
        const int k2 = (i + 2) * 32;
        const int r = tid & 63;
        if (tid < 64) {
          __builtin_prefetch(xrow  + (size_t)r * D_ + k2, 0, 1);
          __builtin_prefetch(wqrow + (size_t)r * D_ + k2, 0, 1);
        } else {
          __builtin_prefetch(wkrow + (size_t)r * D_ + k2, 0, 1);
          __builtin_prefetch(wvrow + (size_t)r * D_ + k2, 0, 1);
        }
      }
    }
    const v16h a = load_a_rm(xs[cur], wave * 16, 0, LDW);
#pragma unroll
    for (int t = 0; t < 4; ++t) {
      accq[t] = wmma16(a, load_b_nm(wqs[cur], t * 16, 0, LDW), accq[t]);
      acck[t] = wmma16(a, load_b_nm(wks[cur], t * 16, 0, LDW), acck[t]);
      accv[t] = wmma16(a, load_b_nm(wvs[cur], t * 16, 0, LDW), accv[t]);
    }
  }

  // ---- +bias, stage q/k row-major, v transposed ----
  wait_stage();              // nothing outstanding may cross the phase change
  __syncthreads();
  {
    const int colb = lane & 15;
    const int rofs = (lane >> 4) * 8;
#pragma unroll
    for (int t = 0; t < 4; ++t) {
      const int col = t * 16 + colb;
      const float bqv = bq[h * DH_ + col];
      const float bkv = bk[h * DH_ + col];
      const float bvv = bv[h * DH_ + col];
#pragma unroll
      for (int r = 0; r < 8; ++r) {
        const int row = wave * 16 + rofs + r;
        qs [row * LDS2 + col] = (_Float16)(accq[t][r] + bqv);
        ks [row * LDS2 + col] = (_Float16)(acck[t][r] + bkv);
        vsT[col * LDS2 + row] = (_Float16)(accv[t][r] + bvv);
      }
    }
  }
  __syncthreads();

  // ---- GEMM 2: qf = q*theta, kf = k*theta (K = 64) ----
  // theta is symmetric, so th[e*64+d] == theta[d][e]: n-major B loader is exact.
  v8f aq[4], ak[4];
#pragma unroll
  for (int t = 0; t < 4; ++t) { aq[t] = {}; ak[t] = {}; }
#pragma unroll
  for (int k0 = 0; k0 < DH_; k0 += 32) {
    const v16h a_q = load_a_rm(qs, wave * 16, k0, LDS2);
    const v16h a_k = load_a_rm(ks, wave * 16, k0, LDS2);
#pragma unroll
    for (int t = 0; t < 4; ++t) {
      const v16h bb = load_b_nm(th, t * 16, k0, DH_);
      aq[t] = wmma16(a_q, bb, aq[t]);
      ak[t] = wmma16(a_k, bb, ak[t]);
    }
  }

  // qf -> global (f16); kf -> LDS transposed [d][n] for the kv reduction
  {
    const int colb = lane & 15;
    const int rofs = (lane >> 4) * 8;
#pragma unroll
    for (int t = 0; t < 4; ++t) {
      const int col = t * 16 + colb;
#pragma unroll
      for (int r = 0; r < 8; ++r) {
        const int row = wave * 16 + rofs + r;
        qf_out[((size_t)bh * N_ + n0 + row) * DH_ + col] = (_Float16)aq[t][r];
        kfsT[col * LDS2 + row] = (_Float16)ak[t][r];
      }
    }
  }
  __syncthreads();

  // ---- GEMM 3: kv_partial[d][e] = sum_n kf[n][d] * v[n][e] (K = 64 rows) ----
  // A[d][n] = kfsT row-major; B[n][e] via vsT n-major ([e][n]).
  v8f akv[4];
#pragma unroll
  for (int t = 0; t < 4; ++t) akv[t] = {};
#pragma unroll
  for (int k0 = 0; k0 < 64; k0 += 32) {
    const v16h a_ = load_a_rm(kfsT, wave * 16, k0, LDS2);
#pragma unroll
    for (int t = 0; t < 4; ++t)
      akv[t] = wmma16(a_, load_b_nm(vsT, t * 16, k0, LDS2), akv[t]);
  }
  {
    float* kvp = kv + (size_t)bh * DH_ * DH_;
    const int colb = lane & 15;
    const int rofs = (lane >> 4) * 8;
#pragma unroll
    for (int t = 0; t < 4; ++t) {
      const int e = t * 16 + colb;
#pragma unroll
      for (int r = 0; r < 8; ++r) {
        const int d = wave * 16 + rofs + r;
        atomicAdd(&kvp[d * DH_ + e], akv[t][r]);
      }
    }
  }
  // ksum partial: kfsT row d is contiguous over n
  if (tid < DH_) {
    float s = 0.0f;
    for (int n = 0; n < 64; ++n) s += (float)kfsT[tid * LDS2 + n];
    atomicAdd(&ksum[(size_t)bh * DH_ + tid], s);
  }
}

// ---------------- Kernel: out = (qf * kv) / (qf . ksum + eps) ----------------
__global__ __launch_bounds__(128) void hh_pass2(
    const _Float16* __restrict__ qf,  // [B*H][N][DH]
    const float* __restrict__ kv,     // [B*H][DH][DH]
    const float* __restrict__ ksum,   // [B*H][DH]
    float* __restrict__ out)          // [B][N][D]
{
  constexpr int LDS2 = 72;
  __shared__ __align__(16) _Float16 qfs [64 * LDS2]; // [row][d]
  __shared__ __align__(16) _Float16 kvsT[64 * LDS2]; // [e][d] (transposed; n-major B)
  __shared__ float ksl[DH_];
  __shared__ float zrow[64];

  const int tid  = threadIdx.x;
  const int wave = tid >> 5;
  const int lane = tid & 31;
  const int bh = blockIdx.y;
  const int h  = bh & (H_ - 1);
  const int b  = bh >> 4;
  const int n0 = blockIdx.x * 64;

  for (int i = tid; i < DH_ * DH_; i += 128) {
    const int d = i >> 6, e = i & 63;
    kvsT[e * LDS2 + d] = (_Float16)kv[(size_t)bh * DH_ * DH_ + i];
  }
  for (int i = tid; i < 64 * DH_; i += 128) {
    const int r = i >> 6, c = i & 63;
    qfs[r * LDS2 + c] = qf[((size_t)bh * N_ + n0 + r) * DH_ + c];
  }
  if (tid < DH_) ksl[tid] = ksum[(size_t)bh * DH_ + tid];
  __syncthreads();

  if (tid < 64) {
    float s = 0.0f;
    for (int d = 0; d < DH_; ++d) s += (float)qfs[tid * LDS2 + d] * ksl[d];
    zrow[tid] = s + EPS_;
  }
  __syncthreads();

  v8f acc[4];
#pragma unroll
  for (int t = 0; t < 4; ++t) acc[t] = {};
#pragma unroll
  for (int k0 = 0; k0 < DH_; k0 += 32) {
    const v16h a = load_a_rm(qfs, wave * 16, k0, LDS2);
#pragma unroll
    for (int t = 0; t < 4; ++t)
      acc[t] = wmma16(a, load_b_nm(kvsT, t * 16, k0, LDS2), acc[t]);
  }

  const int colb = lane & 15;
  const int rofs = (lane >> 4) * 8;
#pragma unroll
  for (int t = 0; t < 4; ++t) {
    const int e = t * 16 + colb;
#pragma unroll
    for (int r = 0; r < 8; ++r) {
      const int row = wave * 16 + rofs + r;
      out[((size_t)b * N_ + n0 + row) * D_ + h * DH_ + e] = acc[t][r] / zrow[row];
    }
  }
}

// ---------------- host-side launcher ----------------
extern "C" void kernel_launch(void* const* d_in, const int* in_sizes, int n_in,
                              void* d_out, int out_size, void* d_ws, size_t ws_size,
                              hipStream_t stream) {
  (void)in_sizes; (void)n_in; (void)out_size; (void)ws_size;

  const float* x  = (const float*)d_in[0];
  const float* Wq = (const float*)d_in[1];
  const float* bq = (const float*)d_in[2];
  const float* Wk = (const float*)d_in[3];
  const float* bk = (const float*)d_in[4];
  const float* Wv = (const float*)d_in[5];
  const float* bv = (const float*)d_in[6];
  const float* w  = (const float*)d_in[7];
  float* out = (float*)d_out;

  // workspace layout
  char* ws = (char*)d_ws;
  _Float16* theta = (_Float16*)ws;                                       // 128 KB
  float* kv   = (float*)(ws + (size_t)H_ * DH_ * DH_ * 2);               // 1 MB
  float* ksum = (float*)((char*)kv + (size_t)B_ * H_ * DH_ * DH_ * 4);   // 16 KB
  _Float16* qf  = (_Float16*)((char*)ksum + (size_t)B_ * H_ * DH_ * 4);  // 32 MB
  _Float16* x16 = qf + (size_t)B_ * H_ * N_ * DH_;                       // 32 MB
  _Float16* wq16 = x16 + (size_t)B_ * N_ * D_;                           // 2 MB
  _Float16* wk16 = wq16 + (size_t)D_ * D_;                               // 2 MB
  _Float16* wv16 = wk16 + (size_t)D_ * D_;                               // 2 MB

  // one-time f32 -> f16 copies (read once from HBM; pass1 then re-reads f16 from L2)
  const int nx = B_ * N_ * D_, nw = D_ * D_;
  hh_cvt<<<nx / 4 / 256, 256, 0, stream>>>(x,  x16,  nx);
  hh_cvt<<<nw / 4 / 256, 256, 0, stream>>>(Wq, wq16, nw);
  hh_cvt<<<nw / 4 / 256, 256, 0, stream>>>(Wk, wk16, nw);
  hh_cvt<<<nw / 4 / 256, 256, 0, stream>>>(Wv, wv16, nw);

  hh_theta<<<H_, 256, 0, stream>>>(w, theta);

  const int nz = B_ * H_ * DH_ * DH_ + B_ * H_ * DH_;  // kv + ksum contiguous
  hh_zero<<<(nz + 255) / 256, 256, 0, stream>>>(kv, nz);

  dim3 grid(N_ / 64, B_ * H_);
  hh_pass1<<<grid, 128, 0, stream>>>(x16, wq16, bq, wk16, bk, wv16, bv,
                                     theta, qf, kv, ksum);
  hh_pass2<<<grid, 128, 0, stream>>>(qf, kv, ksum, out);
}
